// AdaptiveLSTMCellWithSSGRes_19473381720314
// MI455X (gfx1250) — matrix operand order repaired and checked by reading, not compiled, not run on hardware
//
#include <hip/hip_runtime.h>

// ---------------------------------------------------------------------------
// Problem sizes (fixed by the reference)
// ---------------------------------------------------------------------------
#define BATCH 4096
#define INF   1024
#define HF    1024
#define NGATE 5
#define NBIG  6144      // 5*H gates + H alpha-hidden, fused along N

// GEMM tiling
#define BM 128
#define BN 128
#define BK 32
#define LDT 40          // padded K-stride (elements) in LDS: 80B rows (TDM-padded)

typedef __attribute__((ext_vector_type(16))) __bf16        v16bf;
typedef __attribute__((ext_vector_type(8)))  __bf16        v8bf;
typedef __attribute__((ext_vector_type(4)))  __bf16        v4bf;
typedef __attribute__((ext_vector_type(8)))  float         v8f;
typedef __attribute__((ext_vector_type(4)))  unsigned int  u32x4;
typedef __attribute__((ext_vector_type(8)))  int           i32x8;
typedef __attribute__((ext_vector_type(4)))  int           i32x4;

static __device__ __forceinline__ __bf16 f2bf(float f) {
    unsigned int u = __float_as_uint(f);
    u += 0x7fffu + ((u >> 16) & 1u);     // round-to-nearest-even
    unsigned short h = (unsigned short)(u >> 16);
    union { unsigned short s; __bf16 b; } cvt;
    cvt.s = h;
    return cvt.b;
}

static __device__ __forceinline__ v16bf frag_combine(v8bf lo, v8bf hi) {
    v16bf r;
#pragma unroll
    for (int i = 0; i < 8; ++i) { r[i] = lo[i]; r[i + 8] = hi[i]; }
    return r;
}

static __device__ __forceinline__ float sigmoidf(float x) {
    return 1.0f / (1.0f + __expf(-x));
}

// ---------------------------------------------------------------------------
// Tensor Data Mover: DMA one 2D bf16 tile (width x height, row stride in
// elements) from global memory into LDS, padding each 64B row out to 80B so
// LDS rows land at LDT=40 elements.  Descriptor per CDNA5 ISA D# layout.
//   pad_interval = 3  -> pad after every 16 dwords (= 32 bf16 = one K-row)
//   pad_amount   = 3  -> insert 4 dwords (= 8 bf16) of padding
// ---------------------------------------------------------------------------
static __device__ __forceinline__ void tdm_load_tile(
    unsigned lds_byte_addr, const __bf16* gptr,
    int width_elems, int height, int row_stride_elems)
{
    unsigned long long ga = (unsigned long long)(uintptr_t)gptr;
    u32x4 g0;
    g0[0] = 1u;                                          // count=1, user mode
    g0[1] = lds_byte_addr;                               // D#.lds_addr
    g0[2] = (unsigned)(ga & 0xffffffffu);                // global_addr[31:0]
    g0[3] = (unsigned)((ga >> 32) & 0x01ffffffu)         // global_addr[56:32]
          | (2u << 30);                                  // type = 2 (image)
    i32x8 g1;
    g1[0] = (1 << 16)                                    // data_size = 2 bytes
          | (1 << 20)                                    // pad_enable
          | (3 << 22)                                    // pad_interval (16 dw)
          | (3 << 25);                                   // pad_amount   (4 dw)
    g1[1] = (width_elems & 0xffff) << 16;                // tensor_dim0[15:0]
    g1[2] = (((unsigned)width_elems >> 16) & 0xffff)     // tensor_dim0[31:16]
          | ((height & 0xffff) << 16);                   // tensor_dim1[15:0]
    g1[3] = (((unsigned)height >> 16) & 0xffff)          // tensor_dim1[31:16]
          | ((width_elems & 0xffff) << 16);              // tile_dim0
    g1[4] = (height & 0xffff);                           // tile_dim1 (tile_dim2=0)
    g1[5] = row_stride_elems;                            // tensor_dim0_stride lo32
    g1[6] = 0;
    g1[7] = 0;
    i32x4 z4 = {0, 0, 0, 0};
    i32x8 z8 = {0, 0, 0, 0, 0, 0, 0, 0};
    __builtin_amdgcn_tensor_load_to_lds(g0, g1, z4, z4, z8, 0);
}

// ---------------------------------------------------------------------------
// Generic fused GEMM:  C[M,N] = act( A0·W0^T (+ A1·W1^T) + bias )
// A: [M,K] bf16 row-major, W: [N,K] bf16 row-major, row stride == K == 1024.
// act: 0 = none, 1 = relu, 2 = split mode (cols < actSplit: bias1+bias2,
//       no act; cols >= actSplit: biasAlt[col-actSplit] + relu).
// Tiles move global->LDS via TDM (tensor_load_to_lds / s_wait_tensorcnt),
// math via v_wmma_f32_16x16x32_bf16, wave32, 8 waves per 128x128 block.
// ---------------------------------------------------------------------------
__global__ __launch_bounds__(256)
void gemm_bf16_wmma(const __bf16* __restrict__ A0,
                    const __bf16* __restrict__ W0,
                    const __bf16* __restrict__ A1,
                    const __bf16* __restrict__ W1,
                    int K, int nparts,
                    const float* __restrict__ bias1,
                    const float* __restrict__ bias2,
                    const float* __restrict__ biasAlt,
                    int act, int actSplit,
                    float* __restrict__ outF,
                    __bf16* __restrict__ outB,
                    int ldc)
{
    __shared__ alignas(16) __bf16 As[2][BM * LDT];
    __shared__ alignas(16) __bf16 Bs[2][BN * LDT];

    const int tid   = threadIdx.x;
    const int lane  = tid & 31;
    const int wave  = tid >> 5;
    const int waveM = wave & 1;      // 2 waves along M  (64 rows each)
    const int waveN = wave >> 1;     // 4 waves along N  (32 cols each)
    const int tileM = blockIdx.y * BM;
    const int tileN = blockIdx.x * BN;
    const int rlane = lane & 15;

    const unsigned asBase0 = (unsigned)(uintptr_t)(void*)&As[0][0];
    const unsigned asBase1 = (unsigned)(uintptr_t)(void*)&As[1][0];
    const unsigned bsBase0 = (unsigned)(uintptr_t)(void*)&Bs[0][0];
    const unsigned bsBase1 = (unsigned)(uintptr_t)(void*)&Bs[1][0];

    v8f acc[4][2];
#pragma unroll
    for (int mi = 0; mi < 4; ++mi)
#pragma unroll
        for (int ni = 0; ni < 2; ++ni)
#pragma unroll
            for (int r = 0; r < 8; ++r) acc[mi][ni][r] = 0.0f;

    const int steps_per_part = K / BK;
    const int total = steps_per_part * nparts;

    // ---- prologue: DMA tile 0 into buffer 0 ----
    if (wave == 0) {
        tdm_load_tile(asBase0, A0 + (size_t)tileM * K, BK, BM, K);
    } else if (wave == 1) {
        tdm_load_tile(bsBase0, W0 + (size_t)tileN * K, BK, BN, K);
    }
    __builtin_amdgcn_s_wait_tensorcnt((short)0);
    __syncthreads();

    const int khalfA = (lane >> 4) << 3;    // A: lanes 16-31 start at K=8
    const int khalfB = (lane >> 4) << 4;    // unused: B also reads at khalfA (lanes 16-31 at K=8)

    for (int s = 0; s < total; ++s) {
        const int buf   = s & 1;
        const bool more = (s + 1 < total);

        // ---- kick off DMA for the next tile (other buffer) ----
        if (more) {
            int s1   = s + 1;
            int part = s1 / steps_per_part;
            int kt   = (s1 - part * steps_per_part) * BK;
            if (wave == 0) {
                const __bf16* A = part ? A1 : A0;
                tdm_load_tile(buf ? asBase0 : asBase1,
                              A + (size_t)tileM * K + kt, BK, BM, K);
            } else if (wave == 1) {
                const __bf16* W = part ? W1 : W0;
                tdm_load_tile(buf ? bsBase0 : bsBase1,
                              W + (size_t)tileN * K + kt, BK, BN, K);
            }
        }

        // ---- load fragments from LDS (ISA 16-bit A/B lane layouts) ----
        v16bf af[4];
#pragma unroll
        for (int mi = 0; mi < 4; ++mi) {
            int row = waveM * 64 + mi * 16 + rlane;
            const __bf16* p = &As[buf][row * LDT];
            v8bf lo = *(const v8bf*)(const void*)(p + khalfA);
            v8bf hi = *(const v8bf*)(const void*)(p + 16 + khalfA);
            af[mi] = frag_combine(lo, hi);
        }
        v16bf bfr[2];
#pragma unroll
        for (int ni = 0; ni < 2; ++ni) {
            int nn = waveN * 32 + ni * 16 + rlane;
            const __bf16* p = &Bs[buf][nn * LDT + khalfA];
            v8bf lo = *(const v8bf*)(const void*)(p);
            v8bf hi = *(const v8bf*)(const void*)(p + 16);
            bfr[ni] = frag_combine(lo, hi);
        }

        // ---- 8 WMMAs per wave per K-step ----
#pragma unroll
        for (int mi = 0; mi < 4; ++mi)
#pragma unroll
            for (int ni = 0; ni < 2; ++ni)
                acc[mi][ni] = __builtin_amdgcn_wmma_f32_16x16x32_bf16(
                    false, af[mi], false, bfr[ni], (short)0, acc[mi][ni], false, false);

        if (more) {
            __builtin_amdgcn_s_wait_tensorcnt((short)0);
            __syncthreads();
        }
    }

    // ---- epilogue: bias -> act -> store f32 / bf16 ----
#pragma unroll
    for (int ni = 0; ni < 2; ++ni) {
        int col = tileN + waveN * 32 + ni * 16 + rlane;
        bool altRegion = (act == 2) && (col >= actSplit);
        float b;
        if (altRegion) {
            b = biasAlt[col - actSplit];
        } else {
            b = bias1 ? bias1[col] : 0.0f;
            if (bias2) b += bias2[col];
        }
        bool doRelu = (act == 1) || altRegion;
#pragma unroll
        for (int mi = 0; mi < 4; ++mi) {
#pragma unroll
            for (int r = 0; r < 8; ++r) {
                int row = tileM + waveM * 64 + mi * 16 + r + ((lane >> 4) << 3);
                float v = acc[mi][ni][r] + b;
                if (doRelu) v = v > 0.0f ? v : 0.0f;
                size_t idx = (size_t)row * ldc + col;
                if (outF) outF[idx] = v;
                if (outB) outB[idx] = f2bf(v);
            }
        }
    }
}

// ---------------------------------------------------------------------------
// fp32 -> bf16 conversion kernels
// ---------------------------------------------------------------------------
__global__ void cvt4_kernel(const float* __restrict__ in, __bf16* __restrict__ out, int n4)
{
    int i = blockIdx.x * blockDim.x + threadIdx.x;
    if (i >= n4) return;
    float4 f = ((const float4*)in)[i];
    v4bf o;
    o[0] = f2bf(f.x); o[1] = f2bf(f.y); o[2] = f2bf(f.z); o[3] = f2bf(f.w);
    ((v4bf*)out)[i] = o;
}

__global__ void addcvt4_kernel(const float* __restrict__ a, const float* __restrict__ b,
                               __bf16* __restrict__ out, int n4)
{
    int i = blockIdx.x * blockDim.x + threadIdx.x;
    if (i >= n4) return;
    float4 fa = ((const float4*)a)[i];
    float4 fb = ((const float4*)b)[i];
    v4bf o;
    o[0] = f2bf(fa.x + fb.x); o[1] = f2bf(fa.y + fb.y);
    o[2] = f2bf(fa.z + fb.z); o[3] = f2bf(fa.w + fb.w);
    ((v4bf*)out)[i] = o;
}

// strided column-slice pack: dst[r, 0:1024) = bf16(src[r, colOff : colOff+1024))
__global__ void cvtpack4_kernel(const float* __restrict__ src, int srcld, int colOff,
                                __bf16* __restrict__ dst, int rows)
{
    int i = blockIdx.x * blockDim.x + threadIdx.x;   // rows * 256 work items
    int r  = i >> 8;
    int c4 = (i & 255) << 2;
    if (r >= rows) return;
    float4 f = *(const float4*)(src + (size_t)r * srcld + colOff + c4);
    v4bf o;
    o[0] = f2bf(f.x); o[1] = f2bf(f.y); o[2] = f2bf(f.z); o[3] = f2bf(f.w);
    *(v4bf*)(dst + (size_t)r * 1024 + c4) = o;
}

// ---------------------------------------------------------------------------
// alpha layer 2: one wave per row, dot(a_hidden[row,:], a2_w) -> sigmoid
// a_hidden lives in gbuf columns [5120, 6144) with row stride ldg.
// ---------------------------------------------------------------------------
__global__ void alpha_gemv_kernel(const float* __restrict__ ah, int ldg,
                                  const float* __restrict__ a2w,
                                  const float* __restrict__ a2b,
                                  float* __restrict__ alpha, int H)
{
    int row  = blockIdx.x * (blockDim.x >> 5) + (threadIdx.x >> 5);
    int lane = threadIdx.x & 31;
    const float* p = ah + (size_t)row * ldg;
    float s = 0.0f;
    for (int k = lane; k < H; k += 32) s += p[k] * a2w[k];
#pragma unroll
    for (int off = 16; off > 0; off >>= 1) s += __shfl_down(s, off, 32);
    if (lane == 0) alpha[row] = sigmoidf(s + a2b[0]);
}

// ---------------------------------------------------------------------------
// final elementwise: gates -> c_t, h_t   (g has row stride ldg = 6144)
// ---------------------------------------------------------------------------
__global__ void finalize_kernel(const float* __restrict__ g, int ldg,
                                const float* __restrict__ c_prev,
                                const float* __restrict__ alpha,
                                const float* __restrict__ ssg_new,
                                const float* __restrict__ resid,
                                float* __restrict__ h_out,
                                float* __restrict__ c_out,
                                int BH, int H)
{
    int i = blockIdx.x * blockDim.x + threadIdx.x;
    if (i >= BH) return;
    int b = i / H;
    int hcol = i - b * H;
    const float* gr = g + (size_t)b * ldg + hcol;
    float it = sigmoidf(gr[0 * HF]);
    float ft = sigmoidf(gr[1 * HF]);
    float ot = sigmoidf(gr[2 * HF]);
    float ch = tanhf(gr[3 * HF]);
    float st = sigmoidf(gr[4 * HF]);
    float a  = alpha[b];
    float c  = ft * c_prev[i] + it * ch * st * a * ssg_new[i] + resid[i];
    h_out[i] = ot * tanhf(c);
    c_out[i] = c;
}

// ---------------------------------------------------------------------------
// launch
// ---------------------------------------------------------------------------
extern "C" void kernel_launch(void* const* d_in, const int* in_sizes, int n_in,
                              void* d_out, int out_size, void* d_ws, size_t ws_size,
                              hipStream_t stream)
{
    const float* x         = (const float*)d_in[0];
    const float* h_prev    = (const float*)d_in[1];
    const float* c_prev    = (const float*)d_in[2];
    const float* ssg_state = (const float*)d_in[3];
    const float* Wx        = (const float*)d_in[4];
    const float* bWx       = (const float*)d_in[5];
    const float* Ux        = (const float*)d_in[6];
    const float* bUx       = (const float*)d_in[7];
    const float* a1_w      = (const float*)d_in[8];
    const float* a1_b      = (const float*)d_in[9];
    const float* a2_w      = (const float*)d_in[10];
    const float* a2_b      = (const float*)d_in[11];
    const float* r1_w      = (const float*)d_in[12];
    const float* r1_b      = (const float*)d_in[13];
    const float* r2_w      = (const float*)d_in[14];
    const float* r2_b      = (const float*)d_in[15];
    const float* r3_w      = (const float*)d_in[16];
    const float* r3_b      = (const float*)d_in[17];
    const float* ssg_w     = (const float*)d_in[18];
    const float* ssg_b     = (const float*)d_in[19];

    const size_t BH = (size_t)BATCH * HF;

    float* out     = (float*)d_out;
    float* h_out   = out;             // [B,H]
    float* c_out   = out + BH;        // [B,H]
    float* ssg_out = out + 2 * BH;    // [B,H] — written directly by ssg GEMM

    // workspace carve-up (256B aligned)
    char* w = (char*)d_ws;
    size_t off = 0;
    auto alloc = [&](size_t bytes) -> void* {
        void* p = w + off;
        off = (off + bytes + 255) & ~(size_t)255;
        return p;
    };
    __bf16* xb    = (__bf16*)alloc((size_t)BATCH * INF * 2);
    __bf16* hb    = (__bf16*)alloc(BH * 2);
    __bf16* ssb   = (__bf16*)alloc(BH * 2);
    __bf16* r1b   = (__bf16*)alloc(BH * 2);
    __bf16* r2b   = (__bf16*)alloc(BH * 2);
    __bf16* Wcat  = (__bf16*)alloc((size_t)NBIG * INF * 2);   // [6144,1024]
    __bf16* Ucat  = (__bf16*)alloc((size_t)NBIG * HF * 2);    // [6144,1024]
    __bf16* sswb  = (__bf16*)alloc((size_t)HF * HF * 2);
    __bf16* r1wb  = (__bf16*)alloc((size_t)HF * HF * 2);
    __bf16* r2wb  = (__bf16*)alloc((size_t)HF * HF * 2);
    __bf16* r3wb  = (__bf16*)alloc((size_t)HF * HF * 2);
    float*  gbuf  = (float*)alloc((size_t)BATCH * NBIG * 4);  // gates + a_hidden
    float*  resid = (float*)alloc(BH * 4);
    float*  abuf  = (float*)alloc((size_t)BATCH * 4);

    // ---- stage 1: conversions / packing ----
    auto cvt = [&](const float* src, __bf16* dst, size_t n) {
        int n4 = (int)(n / 4);
        cvt4_kernel<<<(n4 + 255) / 256, 256, 0, stream>>>(src, dst, n4);
    };
    cvt(x, xb, (size_t)BATCH * INF);
    cvt(h_prev, hb, BH);
    cvt(Wx, Wcat, (size_t)NGATE * HF * INF);                       // rows 0..5119
    cvt(Ux, Ucat, (size_t)NGATE * HF * HF);                        // rows 0..5119
    cvtpack4_kernel<<<(HF * 256 + 255) / 256, 256, 0, stream>>>(   // rows 5120..6143
        a1_w, INF + HF, 0,   Wcat + (size_t)NGATE * HF * INF, HF);
    cvtpack4_kernel<<<(HF * 256 + 255) / 256, 256, 0, stream>>>(
        a1_w, INF + HF, INF, Ucat + (size_t)NGATE * HF * HF,  HF);
    cvt(ssg_w, sswb, (size_t)HF * HF);
    cvt(r1_w, r1wb, (size_t)HF * HF);
    cvt(r2_w, r2wb, (size_t)HF * HF);
    cvt(r3_w, r3wb, (size_t)HF * HF);
    {
        int n4 = (int)(BH / 4);
        addcvt4_kernel<<<(n4 + 255) / 256, 256, 0, stream>>>(ssg_state, h_prev, ssb, n4);
    }

    // ---- stage 2: GEMMs ----
    dim3 blk(256);
    dim3 gridBig(NBIG / BN, BATCH / BM);   // (48, 32)
    dim3 gridH(HF / BN, BATCH / BM);       // (8, 32)

    // fused: [gates | alpha_hidden] = x·Wcat^T + h·Ucat^T  (+ split biases)
    gemm_bf16_wmma<<<gridBig, blk, 0, stream>>>(
        xb, Wcat, hb, Ucat, 1024, 2,
        bWx, bUx, a1_b,
        2, NGATE * HF,
        gbuf, nullptr, NBIG);

    // ssg_new = (ssg_state + h_prev)·ssg_w^T + ssg_b   (straight into d_out)
    gemm_bf16_wmma<<<gridH, blk, 0, stream>>>(
        ssb, sswb, nullptr, nullptr, 1024, 1,
        ssg_b, nullptr, nullptr, 0, 0,
        ssg_out, nullptr, HF);

    // residual chain
    gemm_bf16_wmma<<<gridH, blk, 0, stream>>>(
        hb, r1wb, nullptr, nullptr, 1024, 1,
        r1_b, nullptr, nullptr, 1, 0,
        nullptr, r1b, HF);
    gemm_bf16_wmma<<<gridH, blk, 0, stream>>>(
        r1b, r2wb, nullptr, nullptr, 1024, 1,
        r2_b, nullptr, nullptr, 1, 0,
        nullptr, r2b, HF);
    gemm_bf16_wmma<<<gridH, blk, 0, stream>>>(
        r2b, r3wb, nullptr, nullptr, 1024, 1,
        r3_b, nullptr, nullptr, 0, 0,
        resid, nullptr, HF);

    // ---- stage 3: alpha GEMV (a_hidden = gbuf cols [5120,6144)) ----
    alpha_gemv_kernel<<<BATCH / 8, 256, 0, stream>>>(
        gbuf + (size_t)NGATE * HF, NBIG, a2_w, a2_b, abuf, HF);

    // ---- stage 4: elementwise finalize ----
    finalize_kernel<<<(int)((BH + 255) / 256), 256, 0, stream>>>(
        gbuf, NBIG, c_prev, abuf, ssg_out, resid, h_out, c_out, (int)BH, HF);

    (void)in_sizes; (void)n_in; (void)out_size; (void)ws_size;
}
